// SparseEncoderVoxelNeXt2DFuse_7370163880455
// MI455X (gfx1250) — matrix operand order repaired and checked by reading, not compiled, not run on hardware
//
#include <hip/hip_runtime.h>
#include <hip/hip_bf16.h>

typedef __attribute__((ext_vector_type(16))) _Float16 v16h;
typedef __attribute__((ext_vector_type(8)))  _Float16 v8h;
typedef __attribute__((ext_vector_type(8)))  float    v8f;

#define EPSF 0.001f
#define CCH 32
#define WFRAG_HALFS (9 * 2 * 32 * 16)     // 9216 halfs = 18432 B per conv layer
#define WFRAG_VEC4  (WFRAG_HALFS / 8)     // 1152 uint4 chunks

// ---------------- rulebook construction ----------------

__global__ void init_map_kernel(int* __restrict__ map, int n) {
  int i = blockIdx.x * blockDim.x + threadIdx.x;
  if (i < n) map[i] = -1;
}

__global__ void zero_pad_kernel(_Float16* __restrict__ z) {
  int i = threadIdx.x;
  if (i < 128) z[i] = (_Float16)0.0f;     // 256 B of zeros for masked gathers
}

__global__ void scatter_map_kernel(int* __restrict__ map,
                                   const int* __restrict__ cb,
                                   const int* __restrict__ cy,
                                   const int* __restrict__ cx,
                                   int n, int Hh, int Ww) {
  int i = blockIdx.x * blockDim.x + threadIdx.x;
  if (i < n) map[(cb[i] * Hh + cy[i]) * Ww + cx[i]] = i;
}

__global__ void build_nidx_kernel(const int* __restrict__ map,
                                  const int* __restrict__ cb,
                                  const int* __restrict__ cy,
                                  const int* __restrict__ cx,
                                  int* __restrict__ nidx,
                                  int n, int Hh, int Ww) {
  int i = blockIdx.x * blockDim.x + threadIdx.x;
  if (i >= n) return;
  int b = cb[i], y = cy[i], x = cx[i];
#pragma unroll
  for (int k = 0; k < 9; ++k) {
    int dy = k / 3 - 1, dx = k % 3 - 1;
    int ny = y + dy, nx = x + dx;
    int j = -1;
    if (ny >= 0 && ny < Hh && nx >= 0 && nx < Ww)
      j = map[(b * Hh + ny) * Ww + nx];   // -1 when voxel empty
    nidx[k * n + i] = j;
  }
}

// ---------------- weight prep: f32 [cv][k][c][o] -> f16 B-fragments ----------------
// wfrag layout: [cv][k][t][lane(32)][i(16)]
//   lane<16 : N=lane,    b[i] = W[k][i   ][o]   (K = 0..15)
//   lane>=16: N=lane-16, b[i] = W[k][16+i][o]   (K = 16..31)
//   o = (lane&15) + 16*t
__global__ void prep_w_kernel(const float* __restrict__ Wk,
                              _Float16* __restrict__ wfrag, int total) {
  int e = blockIdx.x * blockDim.x + threadIdx.x;
  if (e >= total) return;
  int i    = e & 15;
  int lane = (e >> 4) & 31;
  int t    = (e >> 9) & 1;
  int kk   = (e >> 10) % 9;
  int cv   = (e >> 10) / 9;
  int c = ((lane >= 16) ? 16 : 0) + i;
  int o = (lane & 15) + 16 * t;
  wfrag[e] = (_Float16)Wk[(((size_t)cv * 9 + kk) * CCH + c) * CCH + o];
}

__global__ void cvt_kernel(const float* __restrict__ src,
                           float* __restrict__ dst32,
                           _Float16* __restrict__ dst16, int n) {
  int i = blockIdx.x * blockDim.x + threadIdx.x;
  if (i < n) {
    float v = src[i];
    dst32[i] = v;
    dst16[i] = (_Float16)v;
  }
}

// ---------------- gather + WMMA subm-conv + BN (+residual) + ReLU ----------------
// One wave per 16-point tile; 8 waves/block share the 18 KB weight block via LDS.
__global__ __launch_bounds__(256) void subm_conv_wmma_kernel(
    const _Float16* __restrict__ in16,
    const _Float16* __restrict__ zpad,       // >=128 B of zeros (masked gather target)
    const float* __restrict__ identity,      // used when mode==1
    const _Float16* __restrict__ wfrag,      // [9][2][32][16] f16 fragments
    const float* __restrict__ bias, const float* __restrict__ gamma,
    const float* __restrict__ beta, const float* __restrict__ mean,
    const float* __restrict__ var,
    const int* __restrict__ nidx,
    float* __restrict__ out32,               // may be null (skip f32 store)
    _Float16* __restrict__ out16,
    int n, int mode) {
  __shared__ uint4 wlds4[WFRAG_VEC4];        // 18432 B, 16B-aligned

  // cooperative weight staging at B128 granularity: 1152 chunks / 256 threads
  {
    const uint4* srcw = (const uint4*)wfrag;
    for (int i = threadIdx.x; i < WFRAG_VEC4; i += 256) wlds4[i] = srcw[i];
  }
  __syncthreads();                           // before any wave-uniform early exit
  const _Float16* wlds = (const _Float16*)wlds4;

  const int lane = threadIdx.x & 31;
  const int wv   = threadIdx.x >> 5;
  const int tile = blockIdx.x * (blockDim.x >> 5) + wv;
  const int p0   = tile * 16;
  if (p0 >= n) return;                       // wave-uniform: EXEC stays all-ones

  const int r  = lane & 15;
  int p = p0 + r;
  if (p >= n) p = n - 1;                     // safe clamp (N%16==0 in practice)
  const int c0 = (lane < 16) ? 0 : 16;       // A-fragment column base (contiguous K, matches wfrag)

  // hoist all rulebook entries so gather loads can be clustered
  int j[9];
#pragma unroll
  for (int k = 0; k < 9; ++k) j[k] = nidx[k * n + p];

  v8f acc0 = {};
  v8f acc1 = {};

#pragma unroll
  for (int k = 0; k < 9; ++k) {
    // branchless masked gather: select pointer, not data (no EXEC manipulation)
    const _Float16* base = (j[k] >= 0) ? (in16 + (size_t)j[k] * CCH) : zpad;
    union { v16h v; v8h h[2]; } a;
    a.h[0] = *(const v8h*)(base + c0);        // K = c0 .. c0+7
    a.h[1] = *(const v8h*)(base + c0 + 8);    // K = c0+8 .. c0+15
    const _Float16* wb = wlds + ((k * 2) * 32 + lane) * 16;
    v16h b0 = *(const v16h*)(wb);             // t=0: o = 0..15   (ds_load)
    v16h b1 = *(const v16h*)(wb + 32 * 16);   // t=1: o = 16..31  (ds_load)
    acc0 = __builtin_amdgcn_wmma_f32_16x16x32_f16(false, a.v, false, b0,
                                                  (short)0, acc0, false, false);
    acc1 = __builtin_amdgcn_wmma_f32_16x16x32_f16(false, a.v, false, b1,
                                                  (short)0, acc1, false, false);
  }

  // BN folded: y = (x + bias)*sc + sh,  sc = g*rsqrt(var+eps), sh = beta - mean*sc
  const int o0 = lane & 15;
  const int o1 = o0 + 16;
  float sc0 = gamma[o0] * rsqrtf(var[o0] + EPSF);
  float sh0 = beta[o0] - mean[o0] * sc0;
  float bs0 = bias[o0];
  float sc1 = gamma[o1] * rsqrtf(var[o1] + EPSF);
  float sh1 = beta[o1] - mean[o1] * sc1;
  float bs1 = bias[o1];

  const int  mBase = (lane < 16) ? 0 : 8;    // f32 C/D layout: VGPR v -> row v / v+8
  const bool full  = (p0 + 16 <= n);         // wave-uniform: fast path has no guards
#pragma unroll
  for (int v = 0; v < 8; ++v) {
    int pp = p0 + mBase + v;
    if (!full && pp >= n) continue;
    float y0 = (acc0[v] + bs0) * sc0 + sh0;
    float y1 = (acc1[v] + bs1) * sc1 + sh1;
    if (mode == 1) {
      y0 += identity[(size_t)pp * CCH + o0];
      y1 += identity[(size_t)pp * CCH + o1];
    }
    y0 = fmaxf(y0, 0.0f);
    y1 = fmaxf(y1, 0.0f);
    if (out32) {
      out32[(size_t)pp * CCH + o0] = y0;
      out32[(size_t)pp * CCH + o1] = y1;
    }
    out16[(size_t)pp * CCH + o0] = (_Float16)y0;
    out16[(size_t)pp * CCH + o1] = (_Float16)y1;
  }
}

// ---------------- host orchestration ----------------

extern "C" void kernel_launch(void* const* d_in, const int* in_sizes, int n_in,
                              void* d_out, int out_size, void* d_ws, size_t ws_size,
                              hipStream_t stream) {
  const float* feats = (const float*)d_in[0];
  const float* Wk    = (const float*)d_in[1];
  const float* bias  = (const float*)d_in[2];
  const float* gamma = (const float*)d_in[3];
  const float* beta  = (const float*)d_in[4];
  const float* mean  = (const float*)d_in[5];
  const float* var   = (const float*)d_in[6];
  const int*   cb    = (const int*)d_in[7];
  const int*   cy    = (const int*)d_in[8];
  const int*   cx    = (const int*)d_in[9];

  const int N  = in_sizes[7];                // 200000
  const int Bb = 4, Hh = 1440, Ww = 1440, NB = 3;
  const int MAPN = Bb * Hh * Ww;

  char* ws = (char*)d_ws;
  size_t off = 0;
  auto carve = [&](size_t bytes) -> void* {
    void* pv = ws + off;
    off = (off + bytes + 255) & ~(size_t)255;
    return pv;
  };
  int*       idx_map = (int*)carve((size_t)MAPN * 4);
  int*       nidx    = (int*)carve((size_t)9 * N * 4);
  _Float16*  h16     = (_Float16*)carve((size_t)N * CCH * 2);
  _Float16*  mid16   = (_Float16*)carve((size_t)N * CCH * 2);
  _Float16*  wfrag   = (_Float16*)carve((size_t)NB * 2 * WFRAG_HALFS * 2);
  _Float16*  zpad    = (_Float16*)carve(256);
  (void)ws_size; (void)n_in; (void)out_size;

  float* h32 = (float*)d_out;                // running f32 activations live in d_out

  init_map_kernel<<<(MAPN + 255) / 256, 256, 0, stream>>>(idx_map, MAPN);
  zero_pad_kernel<<<1, 128, 0, stream>>>(zpad);
  scatter_map_kernel<<<(N + 255) / 256, 256, 0, stream>>>(idx_map, cb, cy, cx, N, Hh, Ww);
  build_nidx_kernel<<<(N + 255) / 256, 256, 0, stream>>>(idx_map, cb, cy, cx, nidx, N, Hh, Ww);

  const int wtotal = NB * 2 * WFRAG_HALFS;
  prep_w_kernel<<<(wtotal + 255) / 256, 256, 0, stream>>>(Wk, wfrag, wtotal);
  cvt_kernel<<<(N * CCH + 255) / 256, 256, 0, stream>>>(feats, h32, h16, N * CCH);

  const int tiles = (N + 15) / 16;
  const int convBlocks = (tiles + 7) / 8;    // 8 waves (tiles) per 256-thread block
  for (int blk = 0; blk < NB; ++blk) {
    const int cv0 = blk * 2 + 0, cv1 = blk * 2 + 1;
    // conv1: h -> mid (BN + ReLU); f32 copy not needed downstream
    subm_conv_wmma_kernel<<<convBlocks, 256, 0, stream>>>(
        h16, zpad, nullptr, wfrag + (size_t)cv0 * WFRAG_HALFS,
        bias + cv0 * CCH, gamma + cv0 * CCH, beta + cv0 * CCH,
        mean + cv0 * CCH, var + cv0 * CCH,
        nidx, nullptr, mid16, N, 0);
    // conv2: mid -> h (BN + residual(h) + ReLU), updates h32 (d_out) in place
    subm_conv_wmma_kernel<<<convBlocks, 256, 0, stream>>>(
        mid16, zpad, h32, wfrag + (size_t)cv1 * WFRAG_HALFS,
        bias + cv1 * CCH, gamma + cv1 * CCH, beta + cv1 * CCH,
        mean + cv1 * CCH, var + cv1 * CCH,
        nidx, h32, h16, N, 1);
  }
}
